// LSTM_ECG_60971355734107
// MI455X (gfx1250) — matrix operand order repaired and checked by reading, not compiled, not run on hardware
//
#include <hip/hip_runtime.h>
#include <hip/hip_bf16.h>
#include <hip/hip_fp16.h>

typedef _Float16 half_t;
typedef __attribute__((ext_vector_type(16))) _Float16 v16h;
typedef __attribute__((ext_vector_type(8)))  _Float16 v8h;
typedef __attribute__((ext_vector_type(8)))  float    v8f;

#define T_LEN 2500
#define B_SZ  64
#define H_SZ  64
#define G_SZ  256   /* 4*H */

// ---------------------------------------------------------------------------
// WMMA helpers (CDNA5 16x16x32 f16 -> f32)
// ---------------------------------------------------------------------------
__device__ __forceinline__ v8f wmma_f16(v16h a, v16h b, v8f c) {
  // (neg_a, A, neg_b, B, c_mod, C, reuse_a, reuse_b)
  return __builtin_amdgcn_wmma_f32_16x16x32_f16(false, a, false, b, (short)0, c,
                                                false, false);
}

// A tile in contiguous k order (same as load_b_tile): lane L holds row M = L&15.
// lanes 0-15: K = 0..15; lanes 16-31: K = 16..31.
__device__ __forceinline__ v16h load_a_tile(const half_t* __restrict__ src,
                                            int rbase, int kbase, int stride,
                                            int lane) {
  int r  = rbase + (lane & 15);
  int k0 = kbase + ((lane >> 4) << 4);      // +0 or +16
  const half_t* p = src + (size_t)r * stride + k0;
  v8h lo = *(const v8h*)p;
  v8h hi = *(const v8h*)(p + 8);
  v16h a;
#pragma unroll
  for (int i = 0; i < 8; ++i) { a[i] = lo[i]; a[8 + i] = hi[i]; }
  return a;
}

// B tile in contiguous k order (same as load_a_tile): lane L holds column N = L&15;
// lanes 0-15 hold K = kbase+0..15, lanes 16-31 hold K = kbase+16..31.
// == 16 contiguous halves of row-major W[g][k].  32B aligned.
__device__ __forceinline__ v16h load_b_tile(const half_t* __restrict__ W,
                                            int gbase, int kbase, int kstride,
                                            int lane) {
  int g  = gbase + (lane & 15);
  int k0 = kbase + ((lane >> 4) << 4);      // +0 or +16
  return *(const v16h*)(W + (size_t)g * kstride + k0);
}

__device__ __forceinline__ float sigmoid_fast(float x) {
  return 1.0f / (1.0f + __expf(-x));
}
__device__ __forceinline__ float tanh_fast(float x) {
  return 2.0f / (1.0f + __expf(-2.0f * x)) - 1.0f;
}

// CDNA5 async copy: 16B global -> LDS per lane, tracked by ASYNCcnt
__device__ __forceinline__ void async_copy16(unsigned lds_dst, const void* gsrc) {
  asm volatile("global_load_async_to_lds_b128 %0, %1, off"
               :: "v"(lds_dst), "v"(gsrc) : "memory");
}
__device__ __forceinline__ void wait_async_le8() {
  asm volatile("s_wait_asynccnt 0x8" ::: "memory");
}

// ---------------------------------------------------------------------------
// K0: weight preparation (f32 -> f16, bias fusion)
// ---------------------------------------------------------------------------
__global__ __launch_bounds__(256) void prep_weights(
    const float* __restrict__ w_ih0, const float* __restrict__ w_hh0,
    const float* __restrict__ b_ih0, const float* __restrict__ b_hh0,
    const float* __restrict__ w_ih12, const float* __restrict__ w_hh12,
    const float* __restrict__ b_ih12, const float* __restrict__ b_hh12,
    half_t* __restrict__ wih16, half_t* __restrict__ whh16,
    float* __restrict__ bsum) {
  int ld = blockIdx.y;              // layer*2 + dir, 0..5
  int layer = ld >> 1, dir = ld & 1;
  int D = (layer == 0) ? 64 : 128;
  int nIh = G_SZ * D;
  int i = blockIdx.x * 256 + threadIdx.x;
  if (i < nIh) {
    const float* src = (layer == 0)
        ? (w_ih0 + (size_t)dir * G_SZ * 64)
        : (w_ih12 + (size_t)((layer - 1) * 2 + dir) * G_SZ * 128);
    wih16[(size_t)ld * G_SZ * 128 + i] = (half_t)src[i];
  } else if (i < nIh + G_SZ * H_SZ) {
    int k = i - nIh;
    const float* src = (layer == 0)
        ? (w_hh0 + (size_t)dir * G_SZ * H_SZ)
        : (w_hh12 + (size_t)((layer - 1) * 2 + dir) * G_SZ * H_SZ);
    whh16[(size_t)ld * G_SZ * H_SZ + k] = (half_t)src[k];
  } else if (i < nIh + G_SZ * H_SZ + G_SZ) {
    int g = i - nIh - G_SZ * H_SZ;
    const float* bi = (layer == 0) ? (b_ih0 + dir * G_SZ)
                                   : (b_ih12 + ((layer - 1) * 2 + dir) * G_SZ);
    const float* bh = (layer == 0) ? (b_hh0 + dir * G_SZ)
                                   : (b_hh12 + ((layer - 1) * 2 + dir) * G_SZ);
    bsum[ld * G_SZ + g] = bi[g] + bh[g];
  }
}

// ---------------------------------------------------------------------------
// K1: input projection x[B,C,T] -> seq0[T,B,64] (f16).  K=12, VALU is fine.
// ---------------------------------------------------------------------------
__global__ __launch_bounds__(256) void proj_kernel(
    const float* __restrict__ x, const float* __restrict__ Wp,
    const float* __restrict__ bp, half_t* __restrict__ seq0) {
  __shared__ float Ws[64 * 12];
  __shared__ float bs[64];
  int tid = threadIdx.x;
  if (tid < 64) bs[tid] = bp[tid];
  for (int i = tid; i < 64 * 12; i += 256) Ws[i] = Wp[i];
  __syncthreads();
  int t = blockIdx.x * 256 + tid;
  int b = blockIdx.y;
  if (t >= T_LEN) return;
  float xv[12];
#pragma unroll
  for (int c = 0; c < 12; ++c) xv[c] = x[((size_t)b * 12 + c) * T_LEN + t];
  half_t* dst = seq0 + ((size_t)t * B_SZ + b) * 64;
  for (int hh = 0; hh < 64; ++hh) {
    float acc = bs[hh];
#pragma unroll
    for (int c = 0; c < 12; ++c) acc += xv[c] * Ws[hh * 12 + c];
    dst[hh] = (half_t)acc;
  }
}

// ---------------------------------------------------------------------------
// K2: xw = seq @ Wih^T + (bih+bhh), f16 out.  M = T*B, N = 256, K = D.
// ---------------------------------------------------------------------------
__global__ __launch_bounds__(256) void gemm_xw(
    const half_t* __restrict__ seqIn, const half_t* __restrict__ wih16,
    const float* __restrict__ bsum, half_t* __restrict__ xw,
    int layer, int D) {
  int dir = blockIdx.y;
  int ld = layer * 2 + dir;
  const half_t* W = wih16 + (size_t)ld * G_SZ * 128;
  const float* bias = bsum + ld * G_SZ;
  half_t* xwD = xw + (size_t)dir * T_LEN * B_SZ * G_SZ;

  int tid = threadIdx.x, lane = tid & 31, w = tid >> 5;
  int mbase = (blockIdx.x * 4 + (w >> 1)) * 16;
  int gbase = (w & 1) * 128;

  v8f acc[8];
#pragma unroll
  for (int n = 0; n < 8; ++n) acc[n] = (v8f){};

  int KB = D >> 5;
  for (int kb = 0; kb < KB; ++kb) {
    v16h a = load_a_tile(seqIn, mbase, kb * 32, D, lane);
#pragma unroll
    for (int n = 0; n < 8; ++n) {
      v16h b = load_b_tile(W, gbase + n * 16, kb * 32, D, lane);
      acc[n] = wmma_f16(a, b, acc[n]);
    }
  }
  int rb = mbase + ((lane >> 4) << 3);
#pragma unroll
  for (int n = 0; n < 8; ++n) {
    int col = gbase + n * 16 + (lane & 15);
    float bb = bias[col];
#pragma unroll
    for (int r = 0; r < 8; ++r)
      xwD[(size_t)(rb + r) * G_SZ + col] = (half_t)(acc[n][r] + bb);
  }
}

// ---------------------------------------------------------------------------
// K3: persistent recurrent scan.
// grid = (4 batch-groups, 2 directions), 128 threads (4 waves) per WG.
// Wave w owns gate-column set {w, w+4, w+8, w+12} (16 cols each of i,f,g,o)
// -> its WMMA accumulators hold i/f/g/o for the SAME (row,col) in the same
//    lane, so the whole cell update is register-resident; ONE barrier/step.
// h double-buffered in LDS (f16); c in registers (f32).
// xw staged into LDS through a 4-deep global_load_async_to_lds_b128 pipeline.
// ---------------------------------------------------------------------------
__global__ __launch_bounds__(128, 1) void lstm_scan(
    const half_t* __restrict__ xw, const half_t* __restrict__ whh16,
    half_t* __restrict__ seqOut, float* __restrict__ finalBuf,
    int layer, int writeFinal) {
  __shared__ half_t hS[2][16 * H_SZ];       // 2 x 2 KB, [buf][row][j]
  __shared__ half_t xwS[4][16 * G_SZ];      // 4 x 8 KB async-staged xw slices

  int bg  = blockIdx.x;                     // batch group: rows [16bg, 16bg+16)
  int dir = blockIdx.y;
  int tid = threadIdx.x, lane = tid & 31, w = tid >> 5;   // w in 0..3
  const half_t* Whh = whh16 + (size_t)(layer * 2 + dir) * G_SZ * H_SZ;
  const half_t* xwD = xw + (size_t)dir * T_LEN * B_SZ * G_SZ;

  // Static recurrent weights: wave w -> gate tiles {w, w+4, w+8, w+12}
  v16h bt[4][2];
#pragma unroll
  for (int s = 0; s < 4; ++s)
#pragma unroll
    for (int kb = 0; kb < 2; ++kb)
      bt[s][kb] = load_b_tile(Whh, (s * 4 + w) * 16, kb * 32, H_SZ, lane);

  for (int i = tid; i < 16 * H_SZ; i += 128) hS[0][i] = (half_t)0.0f;

  unsigned xwS_base = (unsigned)(uintptr_t)(&xwS[0][0]);
  // Prologue: async-stage xw for steps 0,1,2
#pragma unroll
  for (int p = 0; p < 3; ++p) {
    int tp = dir ? (T_LEN - 1 - p) : p;
    const char* src =
        (const char*)(xwD + ((size_t)tp * B_SZ + bg * 16) * G_SZ) + tid * 64;
    unsigned dst = xwS_base + (unsigned)(p * 8192 + tid * 64);
#pragma unroll
    for (int k = 0; k < 4; ++k) async_copy16(dst + k * 16, src + k * 16);
  }
  wait_async_le8();                 // group 0 complete (2 groups after it)
  __syncthreads();

  float c[8];
#pragma unroll
  for (int r = 0; r < 8; ++r) c[r] = 0.0f;

  int jloc = lane & 15;
  int rowh = (lane >> 4) << 3;      // 0 or 8
  int j    = w * 16 + jloc;         // hidden column within 0..63

  for (int step = 0; step < T_LEN; ++step) {
    int t = dir ? (T_LEN - 1 - step) : step;

    // issue async stage for step+3 (clamped reload at tail keeps cadence)
    {
      int sp = step + 3;
      int tp = (sp < T_LEN) ? sp : (T_LEN - 1);
      int ts = dir ? (T_LEN - 1 - tp) : tp;
      const char* src =
          (const char*)(xwD + ((size_t)ts * B_SZ + bg * 16) * G_SZ) + tid * 64;
      unsigned dst = xwS_base + (unsigned)((sp & 3) * 8192 + tid * 64);
#pragma unroll
      for (int k = 0; k < 4; ++k) async_copy16(dst + k * 16, src + k * 16);
    }

    // ---- recurrent matmul: gates += h(t-1) @ Whh^T, register resident ----
    const half_t* hr = &hS[step & 1][0];
    v16h a0 = load_a_tile(hr, 0, 0,  H_SZ, lane);
    v16h a1 = load_a_tile(hr, 0, 32, H_SZ, lane);
    v8f acc[4];
#pragma unroll
    for (int s = 0; s < 4; ++s) {
      v8f z = (v8f){};
      z = wmma_f16(a0, bt[s][0], z);
      acc[s] = wmma_f16(a1, bt[s][1], z);
    }

    // ---- fused gates + cell update (acc[0..3] = i,f,g,o pre-activations) --
    const half_t* xb = &xwS[step & 3][0];
    half_t* hw = &hS[(step + 1) & 1][0];
    half_t* so = seqOut + ((size_t)t * B_SZ + bg * 16) * 128 + dir * 64;
#pragma unroll
    for (int r = 0; r < 8; ++r) {
      int row = rowh + r;
      int rb = row * G_SZ;
      float gi = acc[0][r] + (float)xb[rb + j];
      float gf = acc[1][r] + (float)xb[rb + 64 + j];
      float gg = acc[2][r] + (float)xb[rb + 128 + j];
      float go = acc[3][r] + (float)xb[rb + 192 + j];
      float iv = sigmoid_fast(gi);
      float fv = sigmoid_fast(gf);
      float gv = tanh_fast(gg);
      float ov = sigmoid_fast(go);
      c[r] = fv * c[r] + iv * gv;
      float hv = ov * tanh_fast(c[r]);
      hw[row * H_SZ + j] = (half_t)hv;
      so[(size_t)row * 128 + j] = (half_t)hv;
    }

    wait_async_le8();   // step+1 slice resident (own groups; barrier -> all)
    __syncthreads();    // h(t) visible to all waves; xwS[step+1] visible
  }

  if (writeFinal) {     // final hidden -> finalBuf[b][dir*64 + j], f32
    const half_t* hf = &hS[T_LEN & 1][0];
    for (int i = tid; i < 16 * H_SZ; i += 128) {
      int row = i >> 6, jj = i & 63;
      finalBuf[(bg * 16 + row) * 128 + dir * 64 + jj] = (float)hf[row * H_SZ + jj];
    }
  }
}

// ---------------------------------------------------------------------------
// K4: tiny fully-connected layers (f32), negligible cost.
// ---------------------------------------------------------------------------
__global__ __launch_bounds__(256) void fc_kernel(
    const float* __restrict__ in, const float* __restrict__ W,
    const float* __restrict__ bias, float* __restrict__ out,
    int Bn, int K, int N, int doRelu) {
  int idx = blockIdx.x * 256 + threadIdx.x;
  if (idx >= Bn * N) return;
  int b = idx / N, j = idx - b * N;
  float acc = bias[j];
  for (int k = 0; k < K; ++k) acc += in[(size_t)b * K + k] * W[(size_t)j * K + k];
  if (doRelu) acc = fmaxf(acc, 0.0f);
  out[(size_t)b * N + j] = acc;
}

// ---------------------------------------------------------------------------
// Workspace layout
// ---------------------------------------------------------------------------
static const size_t SEQ_BYTES = (size_t)T_LEN * B_SZ * 128 * 2;       // 40.96 MB
static const size_t XW_BYTES  = (size_t)2 * T_LEN * B_SZ * G_SZ * 2;  // 163.8 MB
static const size_t OFF_SEQA  = 0;
static const size_t OFF_SEQB  = OFF_SEQA + SEQ_BYTES;
static const size_t OFF_XW    = OFF_SEQB + SEQ_BYTES;
static const size_t OFF_WIH   = OFF_XW + XW_BYTES;
static const size_t OFF_WHH   = OFF_WIH + (size_t)6 * G_SZ * 128 * 2;
static const size_t OFF_BSUM  = OFF_WHH + (size_t)6 * G_SZ * H_SZ * 2;
static const size_t OFF_FINAL = OFF_BSUM + (size_t)6 * G_SZ * 4;
static const size_t OFF_Z1    = OFF_FINAL + (size_t)64 * 128 * 4;
static const size_t OFF_Z2    = OFF_Z1 + (size_t)64 * 128 * 4;

extern "C" void kernel_launch(void* const* d_in, const int* in_sizes, int n_in,
                              void* d_out, int out_size, void* d_ws,
                              size_t ws_size, hipStream_t stream) {
  const float* x      = (const float*)d_in[0];
  const float* Wp     = (const float*)d_in[1];
  const float* bp     = (const float*)d_in[2];
  const float* w_ih0  = (const float*)d_in[3];
  const float* w_hh0  = (const float*)d_in[4];
  const float* b_ih0  = (const float*)d_in[5];
  const float* b_hh0  = (const float*)d_in[6];
  const float* w_ih12 = (const float*)d_in[7];
  const float* w_hh12 = (const float*)d_in[8];
  const float* b_ih12 = (const float*)d_in[9];
  const float* b_hh12 = (const float*)d_in[10];
  const float* Wc1    = (const float*)d_in[11];
  const float* bc1    = (const float*)d_in[12];
  const float* Wc2    = (const float*)d_in[13];
  const float* bc2    = (const float*)d_in[14];
  const float* Wc3    = (const float*)d_in[15];
  const float* bc3    = (const float*)d_in[16];

  char* ws = (char*)d_ws;
  half_t* seqA  = (half_t*)(ws + OFF_SEQA);
  half_t* seqB  = (half_t*)(ws + OFF_SEQB);
  half_t* xwBuf = (half_t*)(ws + OFF_XW);
  half_t* wih16 = (half_t*)(ws + OFF_WIH);
  half_t* whh16 = (half_t*)(ws + OFF_WHH);
  float*  bsum  = (float*)(ws + OFF_BSUM);
  float*  finB  = (float*)(ws + OFF_FINAL);
  float*  z1    = (float*)(ws + OFF_Z1);
  float*  z2    = (float*)(ws + OFF_Z2);

  prep_weights<<<dim3(194, 6), 256, 0, stream>>>(
      w_ih0, w_hh0, b_ih0, b_hh0, w_ih12, w_hh12, b_ih12, b_hh12,
      wih16, whh16, bsum);

  proj_kernel<<<dim3((T_LEN + 255) / 256, B_SZ), 256, 0, stream>>>(x, Wp, bp,
                                                                   seqA);

  // layer 0: in seqA (D=64) -> out seqB
  gemm_xw<<<dim3(2500, 2), 256, 0, stream>>>(seqA, wih16, bsum, xwBuf, 0, 64);
  lstm_scan<<<dim3(4, 2), 128, 0, stream>>>(xwBuf, whh16, seqB, finB, 0, 0);

  // layer 1: in seqB (D=128) -> out seqA
  gemm_xw<<<dim3(2500, 2), 256, 0, stream>>>(seqB, wih16, bsum, xwBuf, 1, 128);
  lstm_scan<<<dim3(4, 2), 128, 0, stream>>>(xwBuf, whh16, seqA, finB, 1, 0);

  // layer 2: in seqA (D=128) -> out seqB, write final hidden
  gemm_xw<<<dim3(2500, 2), 256, 0, stream>>>(seqA, wih16, bsum, xwBuf, 2, 128);
  lstm_scan<<<dim3(4, 2), 128, 0, stream>>>(xwBuf, whh16, seqB, finB, 2, 1);

  // classifier head
  fc_kernel<<<dim3((64 * 128 + 255) / 256), 256, 0, stream>>>(finB, Wc1, bc1,
                                                              z1, 64, 128, 128, 1);
  fc_kernel<<<dim3((64 * 64 + 255) / 256), 256, 0, stream>>>(z1, Wc2, bc2, z2,
                                                             64, 128, 64, 1);
  fc_kernel<<<dim3((64 * 20 + 255) / 256), 256, 0, stream>>>(z2, Wc3, bc3,
                                                             (float*)d_out, 64,
                                                             64, 20, 0);
}
